// MultiHeadAttention_64407329571231
// MI455X (gfx1250) — compile-verified, matrix operand order checked by reading, not run on hardware
//
#include <hip/hip_runtime.h>
#include <hip/hip_bf16.h>
#include <math.h>

// Problem constants (B,S,E,H,D from reference)
#define NB 8
#define NS 1024
#define NE 1024
#define NH 16
#define ND 64

typedef __attribute__((ext_vector_type(16))) __bf16 v16bf;
typedef __attribute__((ext_vector_type(8)))  __bf16 v8bf;
typedef __attribute__((ext_vector_type(8)))  float  v8f;
typedef __attribute__((ext_vector_type(4)))  unsigned int v4u;
typedef __attribute__((ext_vector_type(8)))  int v8i;
typedef __attribute__((ext_vector_type(4)))  int v4i;

__device__ __forceinline__ v16bf cat8(v8bf lo, v8bf hi) {
  return __builtin_shufflevector(lo, hi, 0,1,2,3,4,5,6,7,8,9,10,11,12,13,14,15);
}

// Load one 32-wide K-chunk of an A/B WMMA fragment for this lane.
// ISA layout: lane group hi = lane>>4 owns K = hi*8 + e + (e&8) within the chunk,
// i.e. two contiguous 8-element (16B) runs at +hi*8 and +16+hi*8.
__device__ __forceinline__ v16bf load_frag32(const __bf16* chunkBase, int hi) {
  v8bf lo = *(const v8bf*)(chunkBase + hi * 8);
  v8bf hh = *(const v8bf*)(chunkBase + 16 + hi * 8);
  return cat8(lo, hh);
}

__device__ __forceinline__ v8f wmma_bf16(v16bf a, v16bf b, v8f c) {
  return __builtin_amdgcn_wmma_f32_16x16x32_bf16(
      /*neg_a=*/false, a, /*neg_b=*/false, b,
      /*c_mod=*/(short)0, c, /*reuse_a=*/false, /*reuse_b=*/false);
}

__device__ __forceinline__ float gelu_tanh(float x) {
  const float c = 0.7978845608028654f; // sqrt(2/pi)
  float x3 = x * x * x;
  return x * 0.5f * (1.0f + tanhf(c * (x + 0.044715f * x3)));
}

// TDM: load a [16 rows x NS cols] bf16 tile (row stride NS elements) from global
// into LDS at byte offset ldsByteOff. D# per CDNA5 ISA §8 (2D tensor, groups 2/3 unused).
// This toolchain exposes the 6-arg builtin: (v4u, v8i, v4i, v4i, v8i, i32 cpol).
__device__ __forceinline__ void tdm_load_v16rows(const __bf16* gsrc, unsigned ldsByteOff) {
  unsigned long long ga = (unsigned long long)(size_t)gsrc;
  v4u g0;
  g0.x = 1u;                                   // count=1, is_restore=0
  g0.y = ldsByteOff;                           // lds_addr (bytes)
  g0.z = (unsigned)(ga & 0xFFFFFFFFu);         // global_addr[31:0]
  g0.w = (unsigned)((ga >> 32) & 0x1FFFFFFu)   // global_addr[56:32]
         | 0x80000000u;                        // type=2 ("image") in bits[127:126]
  v8i g1;
  g1[0] = (int)(1u << 16);                     // data_size=1 (2 bytes/elem)
  g1[1] = (int)((NS & 0xFFFFu) << 16);         // tensor_dim0[15:0]  @ bits 63:48
  g1[2] = (int)(16u << 16);                    // tensor_dim1[15:0]  @ bits 95:80
  g1[3] = (int)((NS & 0xFFFFu) << 16);         // tile_dim0          @ bits 127:112
  g1[4] = 16;                                  // tile_dim1          @ bits 143:128
  g1[5] = NS;                                  // tensor_dim0_stride low 32
  g1[6] = 0;
  g1[7] = 0;
  v4i z4 = {0, 0, 0, 0};
  v8i z8 = {0, 0, 0, 0, 0, 0, 0, 0};
  __builtin_amdgcn_tensor_load_to_lds(g0, g1, z4, z4, z8, 0);
}

// ---------------- Stage 1: precision conversion ----------------

__global__ void k_f32_to_bf16(const float* __restrict__ src,
                              __bf16* __restrict__ dst, int n) {
  int i = blockIdx.x * 256 + threadIdx.x;
  if (i < n) dst[i] = (__bf16)src[i];
}

// LDS-tiled transpose: W [E,E] fp32 -> Wt [E,E] bf16, Wt[j,i] = W[i,j].
// Coalesced reads AND writes (32x32 tile, pad to kill bank conflicts).
__global__ __launch_bounds__(256)
void k_w_transpose_bf16(const float* __restrict__ w, __bf16* __restrict__ wt) {
  __shared__ float t[32][33];
  const int bx = blockIdx.x * 32;
  const int by = blockIdx.y * 32;
  const int tx = threadIdx.x;       // 0..31
  const int ty = threadIdx.y;       // 0..7
#pragma unroll
  for (int r = 0; r < 32; r += 8)
    t[ty + r][tx] = w[(size_t)(by + ty + r) * NE + bx + tx];
  __syncthreads();
#pragma unroll
  for (int r = 0; r < 32; r += 8)
    wt[(size_t)(bx + ty + r) * NE + by + tx] = (__bf16)t[tx][ty + r];
}

// ---------------- Stage 2: projection GEMM + bias + GELU ----------------
// grid = (E/64, B*S/16, 3), block = 32: one wave computes a 16x64 output slab
// (4 column tiles share one A fragment -> 4x A reuse, 4 WMMA per K chunk).
// z: 0=Q, 1=K, 2=V.  Q,K stored [B,H,S,D] bf16; V stored transposed [B,H,D,S].
__global__ __launch_bounds__(32)
void k_proj_gelu(const __bf16* __restrict__ xq, const __bf16* __restrict__ xk,
                 const __bf16* __restrict__ xv,
                 const __bf16* __restrict__ wqt, const __bf16* __restrict__ wkt,
                 const __bf16* __restrict__ wvt,
                 const float* __restrict__ bq, const float* __restrict__ bk,
                 const float* __restrict__ bv,
                 __bf16* __restrict__ qo, __bf16* __restrict__ ko,
                 __bf16* __restrict__ vo) {
  const int ct = blockIdx.x;            // 64-wide column slab over E
  const int rt = blockIdx.y;            // 16-row tile over B*S
  const int mz = blockIdx.z;
  const __bf16* X  = (mz == 0) ? xq : (mz == 1) ? xk : xv;
  const __bf16* Wt = (mz == 0) ? wqt : (mz == 1) ? wkt : wvt;
  const float*  bias = (mz == 0) ? bq : (mz == 1) ? bk : bv;

  const int lane = threadIdx.x;
  const int l15 = lane & 15;
  const int hi  = lane >> 4;

  const __bf16* xrow = X + (size_t)(rt * 16 + l15) * NE;
  const __bf16* w0 = Wt + (size_t)(ct * 64 +  0 + l15) * NE;
  const __bf16* w1 = Wt + (size_t)(ct * 64 + 16 + l15) * NE;
  const __bf16* w2 = Wt + (size_t)(ct * 64 + 32 + l15) * NE;
  const __bf16* w3 = Wt + (size_t)(ct * 64 + 48 + l15) * NE;

  v8f acc0 = {}, acc1 = {}, acc2 = {}, acc3 = {};
  for (int c = 0; c < NE / 32; ++c) {   // 32 iters x 4 WMMA
    v16bf a = load_frag32(xrow + c * 32, hi);
    acc0 = wmma_bf16(a, load_frag32(w0 + c * 32, hi), acc0);
    acc1 = wmma_bf16(a, load_frag32(w1 + c * 32, hi), acc1);
    acc2 = wmma_bf16(a, load_frag32(w2 + c * 32, hi), acc2);
    acc3 = wmma_bf16(a, load_frag32(w3 + c * 32, hi), acc3);
  }

  const int bidx  = (rt * 16) / NS;     // 16-row tile never straddles batches
  const int sbase = (rt * 16) % NS;

#pragma unroll
  for (int sub = 0; sub < 4; ++sub) {
    const v8f acc = (sub == 0) ? acc0 : (sub == 1) ? acc1 : (sub == 2) ? acc2 : acc3;
    const int ecol = ct * 64 + sub * 16 + l15;   // N fixed per lane
    const float bval = bias[ecol];
    const int h = ecol >> 6;                     // E = H*D, D=64
    const int d = ecol & 63;
#pragma unroll
    for (int r = 0; r < 8; ++r) {
      const int M = r + 8 * hi;                  // C/D layout
      const float g = gelu_tanh(acc[r] + bval);
      if (mz == 0)
        qo[(((size_t)bidx * NH + h) * NS + sbase + M) * ND + d] = (__bf16)g;
      else if (mz == 1)
        ko[(((size_t)bidx * NH + h) * NS + sbase + M) * ND + d] = (__bf16)g;
      else
        vo[(((size_t)bidx * NH + h) * ND + d) * NS + sbase + M] = (__bf16)g; // V^T
    }
  }
}

// ---------------- Stage 3: fused scores -> softmax -> attn@V ----------------
// grid = (S/16, H, B), block = 32 (one wave owns 16 query rows).
// Dynamic LDS (160 KB): [0,64K) f32 score slab; [64K,96K) bf16 probs;
// [96K,160K) 2x32KB double-buffered V^T tiles staged by the Tensor Data Mover.
#define LDS_SC   0u
#define LDS_PROB 65536u
#define LDS_VB0  98304u
#define LDS_VB1  131072u

__global__ __launch_bounds__(32)
void k_attention(const __bf16* __restrict__ qp, const __bf16* __restrict__ kp,
                 const __bf16* __restrict__ vpt,
                 const float* __restrict__ mask,
                 float* __restrict__ concat, float* __restrict__ attn) {
  extern __shared__ char smem[];
  float*  sc   = (float*)(smem + LDS_SC);
  __bf16* prob = (__bf16*)(smem + LDS_PROB);

  const int qt = blockIdx.x;
  const int h  = blockIdx.y;
  const int b  = blockIdx.z;
  const int lane = threadIdx.x;
  const int l15 = lane & 15;
  const int hi  = lane >> 4;

  const size_t headoff = (size_t)b * NH + h;
  const __bf16* qhead = qp  + headoff * NS * ND;
  const __bf16* khead = kp  + headoff * NS * ND;
  const __bf16* vhead = vpt + headoff * ND * NS;   // [D, S]
  const int s0 = qt * 16;

  // A operand (q tile) reused across all 64 key tiles: K-dim = D = 64 -> 2 chunks
  const __bf16* qrow = qhead + (size_t)(s0 + l15) * ND;
  const v16bf a0 = load_frag32(qrow, hi);
  const v16bf a1 = load_frag32(qrow + 32, hi);
  const float* mrow = mask + (size_t)b * NS;

  // ---- scores = (q k^T) * 1/sqrt(D) + mask * -1e9 -> LDS ----
  for (int ctile = 0; ctile < NS / 16; ++ctile) {
    const int key = ctile * 16 + l15;              // B-operand column = key row
    const __bf16* krow = khead + (size_t)key * ND; // row-major over D == B layout
    v8f acc = {};
    acc = wmma_bf16(a0, load_frag32(krow, hi), acc);
    acc = wmma_bf16(a1, load_frag32(krow + 32, hi), acc);
    const float mv = mrow[key] * (-1e9f);          // N fixed per lane
#pragma unroll
    for (int r = 0; r < 8; ++r) {
      const int M = r + 8 * hi;
      sc[M * NS + key] = acc[r] * 0.125f + mv;     // 1/sqrt(64)
    }
  }
  // Single wave: per-wave LDS ops complete in order -> no barrier needed.

  // Kick off TDM for the first V^T slab now; its latency hides under softmax.
  tdm_load_v16rows(vhead + 0 * 16 * NS, LDS_VB0);

  // ---- row-wise softmax, 32 lanes per row; write attn to HBM exactly once ----
  for (int m = 0; m < 16; ++m) {
    float mx = -3.0e38f;
    for (int c = lane; c < NS; c += 32) mx = fmaxf(mx, sc[m * NS + c]);
#pragma unroll
    for (int o = 16; o > 0; o >>= 1) mx = fmaxf(mx, __shfl_xor(mx, o, 32));
    float sum = 0.0f;
    for (int c = lane; c < NS; c += 32) {
      float e = expf(sc[m * NS + c] - mx);
      sc[m * NS + c] = e;
      sum += e;
    }
#pragma unroll
    for (int o = 16; o > 0; o >>= 1) sum += __shfl_xor(sum, o, 32);
    const float inv = 1.0f / sum;
    float* arow = attn + (headoff * NS + s0 + m) * NS;
    for (int c = lane; c < NS; c += 32) {          // coalesced store
      float p = sc[m * NS + c] * inv;
      prob[m * NS + c] = (__bf16)p;                // bf16 copy for WMMA A-operand
      arow[c] = p;
    }
  }

  // ---- out[16, 64] = attn[16, S] @ v[S, 64]; V^T staged via TDM, 2 buffers ----
  float* cb = concat + (size_t)b * NS * NE + (size_t)h * NS * ND;
  const __bf16* prow = prob + (size_t)l15 * NS;

  for (int dt = 0; dt < 4; ++dt) {
    if (dt < 3) {
      tdm_load_v16rows(vhead + (size_t)(dt + 1) * 16 * NS,
                       (dt & 1) ? LDS_VB0 : LDS_VB1);
      __builtin_amdgcn_s_wait_tensorcnt(1);        // current buffer ready
    } else {
      __builtin_amdgcn_s_wait_tensorcnt(0);
    }
    const __bf16* vb = (const __bf16*)(smem + ((dt & 1) ? LDS_VB1 : LDS_VB0));
    const __bf16* vrow = vb + (size_t)l15 * NS;    // lane's d-column of this d-tile
    v8f acc = {};
    for (int c = 0; c < NS / 32; ++c) {
      v16bf a = load_frag32(prow + c * 32, hi);    // ds_load_b128 x2
      v16bf bfr = load_frag32(vrow + c * 32, hi);  // ds_load_b128 x2
      acc = wmma_bf16(a, bfr, acc);
    }
    // concat (head-major reshape): flat = b*S*E + h*S*D + s*D + d
#pragma unroll
    for (int r = 0; r < 8; ++r) {
      const int M = r + 8 * hi;
      cb[(size_t)(s0 + M) * ND + dt * 16 + l15] = acc[r];
    }
  }
}

// ---------------- host-side launch ----------------

extern "C" void kernel_launch(void* const* d_in, const int* in_sizes, int n_in,
                              void* d_out, int out_size, void* d_ws, size_t ws_size,
                              hipStream_t stream) {
  (void)in_sizes; (void)n_in; (void)out_size; (void)ws_size;

  const float* Q    = (const float*)d_in[0];
  const float* K    = (const float*)d_in[1];
  const float* V    = (const float*)d_in[2];
  const float* mask = (const float*)d_in[3];
  const float* Wq   = (const float*)d_in[4];
  const float* bq   = (const float*)d_in[5];
  const float* Wk   = (const float*)d_in[6];
  const float* bk   = (const float*)d_in[7];
  const float* Wv   = (const float*)d_in[8];
  const float* bv   = (const float*)d_in[9];

  const size_t NIN = (size_t)NB * NS * NE;   // 8M elements
  const size_t NW  = (size_t)NE * NE;        // 1M elements

  __bf16* ws  = (__bf16*)d_ws;
  __bf16* xq  = ws;
  __bf16* xk  = xq + NIN;
  __bf16* xv  = xk + NIN;
  __bf16* wqt = xv + NIN;
  __bf16* wkt = wqt + NW;
  __bf16* wvt = wkt + NW;
  __bf16* qp  = wvt + NW;
  __bf16* kp  = qp + NIN;                    // B*H*S*D == B*S*E
  __bf16* vpt = kp + NIN;

  float* concat = (float*)d_out;
  float* attn   = concat + NIN;

  // Stage 1: dtype conversion / coalesced LDS-tiled weight transpose
  const int n32 = (int)NIN;
  k_f32_to_bf16<<<(n32 + 255) / 256, 256, 0, stream>>>(Q, xq, n32);
  k_f32_to_bf16<<<(n32 + 255) / 256, 256, 0, stream>>>(K, xk, n32);
  k_f32_to_bf16<<<(n32 + 255) / 256, 256, 0, stream>>>(V, xv, n32);
  {
    dim3 tb(32, 8);
    dim3 tg(NE / 32, NE / 32);
    k_w_transpose_bf16<<<tg, tb, 0, stream>>>(Wq, wqt);
    k_w_transpose_bf16<<<tg, tb, 0, stream>>>(Wk, wkt);
    k_w_transpose_bf16<<<tg, tb, 0, stream>>>(Wv, wvt);
  }

  // Stage 2: fused projection + bias + GELU (bf16 WMMA, f32 accumulate)
  dim3 pg(NE / 64, (NB * NS) / 16, 3);
  k_proj_gelu<<<pg, 32, 0, stream>>>(xq, xk, xv, wqt, wkt, wvt,
                                     bq, bk, bv, qp, kp, vpt);

  // Stage 3: fused scores -> softmax -> attn@V with TDM-staged V tiles
  const int attnLds = 160 * 1024;
  (void)hipFuncSetAttribute((const void*)k_attention,
                            hipFuncAttributeMaxDynamicSharedMemorySize, attnLds);
  dim3 ag(NS / 16, NH, NB);
  k_attention<<<ag, 32, attnLds, stream>>>(qp, kp, vpt, mask, concat, attn);
}
